// TimeEvolutionNuetrino_9929964388493
// MI455X (gfx1250) — matrix operand order repaired and checked by reading, not compiled, not run on hardware
//
#include <hip/hip_runtime.h>
#include <hip/hip_bf16.h>

typedef __attribute__((ext_vector_type(16))) _Float16 v16h;
typedef __attribute__((ext_vector_type(8)))  _Float16 v8h;
typedef __attribute__((ext_vector_type(8)))  float    v8f;
typedef __attribute__((ext_vector_type(2)))  float    v2f;

#define ROWS 32
#define OFF0 0
#define OFF1 (ROWS * 264)   // region0 capacity: max even-layer stride 264

struct Biases { const float* b[7]; };

// ---------------------------------------------------------------------------
// Prep: transpose fp32 W[din][dout] -> f16 WT[dout][dinp], zero-padded K
// ---------------------------------------------------------------------------
__global__ void transpose_to_f16(const float* __restrict__ W, _Float16* __restrict__ WT,
                                 int din, int dinp, int dout) {
  int idx = blockIdx.x * blockDim.x + threadIdx.x;
  if (idx < dout * dinp) {
    int n = idx / dinp, k = idx - n * dinp;
    WT[idx] = (k < din) ? (_Float16)W[k * dout + n] : (_Float16)0.f;
  }
}

// ---------------------------------------------------------------------------
// Fused MLP: 32 samples per block, activations ping-pong in LDS (f16),
// all hidden layers as 16x16 tiles via v_wmma_f32_16x16x32_f16.
// 48-wide layers are K-padded to 64 with zeros (both activations and weights).
// ---------------------------------------------------------------------------
__launch_bounds__(256)
__global__ void mlp_fused(const float* __restrict__ x,
                          const float* __restrict__ W0,
                          const float* __restrict__ b0,
                          const _Float16* __restrict__ wt_base,
                          Biases bs,
                          float* __restrict__ gamma) {
  __shared__ __align__(16) _Float16 lds[OFF1 + ROWS * 520]; // 50,176 bytes
  const int tid  = threadIdx.x;
  const int lane = tid & 31;
  const int wid  = tid >> 5;
  const int a0   = blockIdx.x * ROWS;
  const int m    = lane & 15;
  const int hi   = (lane >> 4) & 1;

  // ---- layer 0: 1 -> 48 (outer product, plain VALU), K-padded to 64 ----
  for (int idx = tid; idx < ROWS * 64; idx += 256) {
    int r = idx >> 6, c = idx & 63;
    float v = 0.f;
    if (c < 48) {
      v = x[a0 + r] * W0[c] + b0[c];
      v = fmaxf(v, 0.01f * v);                 // LeakyReLU, slope in (0,1)
    }
    lds[OFF0 + r * 72 + c] = (_Float16)v;
  }

  const int dims[9]  = {1, 48, 128, 256, 512, 256, 128, 48, 16};
  // K-padded (48->64) transposed-weight offsets in halves
  const int wtoff[8] = {0, 0, 8192, 40960, 172032, 303104, 335872, 342016};

  for (int l = 1; l <= 7; ++l) {
    __syncthreads();
    const int Win   = dims[l],  Wout = dims[l + 1];
    const int KpIn  = (Win  == 48) ? 64 : Win;
    const int KpOut = (Wout == 48) ? 64 : Wout;
    const int Sin   = KpIn + 8, Sout = KpOut + 8;
    const _Float16* inBuf  = lds + (((l - 1) & 1) ? OFF1 : OFF0);
    _Float16*       outBuf = lds + ((l & 1) ? OFF1 : OFF0);
    const _Float16* wt   = wt_base + wtoff[l];
    const float*    bias = bs.b[l - 1];
    const int ntiles = 2 * (Wout >> 4);          // 2 row-bands x Wout/16 col-tiles

    for (int t = wid; t < ntiles; t += 8) {
      const int band = t & 1;
      const int n0   = (t >> 1) << 4;
      v8f acc = {};
      // A-frag: row = band*16 + m; K chunks {base..base+7},{base+16..base+23}, base=hi*8
      const _Float16* arow = inBuf + (band * 16 + m) * Sin + hi * 8;
      // B-frag: output-neuron row n0+m; K chunks {base..base+7},{base+16..base+23}, base=hi*8
      const _Float16* brow = wt + (n0 + m) * KpIn + hi * 8;
      #pragma unroll 2
      for (int k0 = 0; k0 < KpIn; k0 += 32) {
        union { v16h v; v8h h[2]; } AU, BU;
        AU.h[0] = *(const v8h*)(arow + k0);
        AU.h[1] = *(const v8h*)(arow + k0 + 16);
        BU.h[0] = *(const v8h*)(brow + k0);
        BU.h[1] = *(const v8h*)(brow + k0 + 16);
        acc = __builtin_amdgcn_wmma_f32_16x16x32_f16(
            false, AU.v, false, BU.v, (short)0, acc, false, false);
      }
      const float bn = bias[n0 + m];
      if (l < 7) {
        _Float16* op = outBuf + (band * 16 + hi * 8) * Sout + n0 + m;
        #pragma unroll
        for (int r = 0; r < 8; ++r) {
          float v = acc[r] + bn;
          v = fmaxf(v, 0.01f * v);
          op[r * Sout] = (_Float16)v;
        }
      } else {
        // final layer (48 -> 16): no activation; write Gamma fp32
        float* gp = gamma + (size_t)(a0 + band * 16 + hi * 8) * 16 + m;
        #pragma unroll
        for (int r = 0; r < 8; ++r) gp[r * 16] = acc[r] + bn;
      }
    }

    // zero K-pad columns 48..63 when this layer produced a 48-wide activation
    if (Wout == 48) {
      for (int idx = tid; idx < ROWS * 16; idx += 256)
        outBuf[(idx >> 4) * Sout + 48 + (idx & 15)] = (_Float16)0.f;
    }
  }
}

// ---------------------------------------------------------------------------
// expm: complex 8x8 -> real 16x16 block embedding [[Re,-Im],[Im,Re]];
// one wave per sample; Taylor Horner (10) + 8 squarings with
// v_wmma_f32_16x16x4_f32 chained 4x per 16x16x16 matmul.
// ---------------------------------------------------------------------------
__launch_bounds__(256)
__global__ void expm_kernel(const float* __restrict__ gR,
                            const float* __restrict__ gI,
                            const float* __restrict__ H,   // [16][8][8] complex interleaved
                            float* __restrict__ out) {
  __shared__ __align__(16) float mats[8][16 * 18 + 2];
  __shared__ float Hs[2048];                                // HBasis staged once
  const int  tid  = threadIdx.x;
  const int  lane = tid & 31;
  const int  wid  = tid >> 5;
  const long a    = (long)blockIdx.x * 8 + wid;
  const int  m    = lane & 15;
  const int  hi   = (lane >> 4) & 1;
  float* mat = &mats[wid][0];

  for (int idx = tid; idx < 2048; idx += 256) Hs[idx] = H[idx];
  __syncthreads();

  // ---- Ms = (block-embed of sum_b Gamma_b H_b) * 2^-8, built as A-fragments ----
  float gr[16], gi[16];
  #pragma unroll
  for (int b = 0; b < 16; ++b) { gr[b] = gR[a * 16 + b]; gi[b] = gI[a * 16 + b]; }

  v2f msA[4];
  {
    const int  i   = m & 7;
    const bool mlo = (m < 8);
    #pragma unroll
    for (int kc = 0; kc < 4; ++kc) {
      #pragma unroll
      for (int j = 0; j < 2; ++j) {
        const int  c   = kc * 4 + hi * 2 + j;
        const int  jj  = c & 7;
        const bool clo = (c < 8);
        const float* hp = Hs + (i * 8 + jj) * 2;
        float s = 0.f;
        #pragma unroll
        for (int b = 0; b < 16; ++b) {
          float hr = hp[b * 128], him = hp[b * 128 + 1];
          float re = gr[b] * hr - gi[b] * him;
          float im = gr[b] * him + gi[b] * hr;
          s += (mlo == clo) ? re : (mlo ? -im : im);
        }
        msA[kc][j] = s * (1.0f / 256.0f);
      }
    }
  }

  // ---- T = I (D-form: VGPR r -> row r + hi*8, col m) ----
  v8f t;
  #pragma unroll
  for (int r = 0; r < 8; ++r) t[r] = ((r + hi * 8) == m) ? 1.f : 0.f;

  // ---- Horner: T = I + (Ms @ T)/k,  k = 10..1 ----
  for (int k = 10; k >= 1; --k) {
    #pragma unroll
    for (int r = 0; r < 8; ++r) mat[(r + hi * 8) * 18 + m] = t[r];
    asm volatile("s_wait_dscnt 0" ::: "memory");   // wave-internal LDS RAW
    v2f bf[4];
    #pragma unroll
    for (int kc = 0; kc < 4; ++kc) {
      const int kk = kc * 4 + hi * 2;
      bf[kc][0] = mat[kk * 18 + m];
      bf[kc][1] = mat[(kk + 1) * 18 + m];
    }
    v8f acc = {};
    #pragma unroll
    for (int kc = 0; kc < 4; ++kc)
      acc = __builtin_amdgcn_wmma_f32_16x16x4_f32(
          false, msA[kc], false, bf[kc], (short)0, acc, false, false);
    const float invk = 1.f / (float)k;
    #pragma unroll
    for (int r = 0; r < 8; ++r)
      t[r] = (((r + hi * 8) == m) ? 1.f : 0.f) + acc[r] * invk;
  }

  // ---- 8 squarings: T = T @ T ----
  for (int sq = 0; sq < 8; ++sq) {
    #pragma unroll
    for (int r = 0; r < 8; ++r) mat[(r + hi * 8) * 18 + m] = t[r];
    asm volatile("s_wait_dscnt 0" ::: "memory");
    v2f af[4], bf[4];
    #pragma unroll
    for (int kc = 0; kc < 4; ++kc) {
      const int kk = kc * 4 + hi * 2;
      af[kc][0] = mat[m * 18 + kk];
      af[kc][1] = mat[m * 18 + kk + 1];
      bf[kc][0] = mat[kk * 18 + m];
      bf[kc][1] = mat[(kk + 1) * 18 + m];
    }
    v8f acc = {};
    #pragma unroll
    for (int kc = 0; kc < 4; ++kc)
      acc = __builtin_amdgcn_wmma_f32_16x16x4_f32(
          false, af[kc], false, bf[kc], (short)0, acc, false, false);
    t = acc;
  }

  // ---- extract expm(M): Re = T[0:8][0:8] (lanes<16), Im = T[8:16][0:8] ----
  if (m < 8) {
    float* op = out + (a * 64 + m) * 2 + hi;
    #pragma unroll
    for (int r = 0; r < 8; ++r) op[r * 16] = t[r];
  }
}

// ---------------------------------------------------------------------------
// Launch
// ---------------------------------------------------------------------------
extern "C" void kernel_launch(void* const* d_in, const int* in_sizes, int n_in,
                              void* d_out, int out_size, void* d_ws, size_t ws_size,
                              hipStream_t stream) {
  (void)n_in; (void)out_size; (void)ws_size;
  const int A = in_sizes[0];                       // 131072 samples
  const float* x = (const float*)d_in[0];
  // leaf order: x, (W0,b0..W7,b7)_r [1..16], (W0,b0..W7,b7)_i [17..32], HBasis [33]
  const float* H = (const float*)d_in[33];

  char*  ws = (char*)d_ws;
  float* gR = (float*)ws;                          // A*16 f32
  float* gI = gR + (size_t)A * 16;
  _Float16* wtR = (_Float16*)(ws + (size_t)A * 16 * 4 * 2);
  _Float16* wtI = wtR + 344064;                    // 343040 halves per MLP, padded

  static const int dims[9]  = {1, 48, 128, 256, 512, 256, 128, 48, 16};
  static const int wtoff[8] = {0, 0, 8192, 40960, 172032, 303104, 335872, 342016};

  for (int p = 0; p < 2; ++p) {
    _Float16* wt = p ? wtI : wtR;
    const int base = 1 + p * 16;
    for (int l = 1; l <= 7; ++l) {
      const int din  = dims[l];
      const int dinp = (din == 48) ? 64 : din;
      const int dout = dims[l + 1];
      const int tot  = dout * dinp;
      const float* W = (const float*)d_in[base + 2 * l];
      transpose_to_f16<<<(tot + 255) / 256, 256, 0, stream>>>(W, wt + wtoff[l],
                                                              din, dinp, dout);
    }
  }

  for (int p = 0; p < 2; ++p) {
    Biases bs;
    const int base = 1 + p * 16;
    for (int l = 1; l <= 7; ++l) bs.b[l - 1] = (const float*)d_in[base + 2 * l + 1];
    const float* W0 = (const float*)d_in[base + 0];
    const float* b0 = (const float*)d_in[base + 1];
    mlp_fused<<<A / ROWS, 256, 0, stream>>>(x, W0, b0, p ? wtI : wtR, bs,
                                            p ? gI : gR);
  }

  expm_kernel<<<A / 8, 256, 0, stream>>>(gR, gI, H, (float*)d_out);
}
